// UnetGraphSAGE_29326036697882
// MI455X (gfx1250) — matrix operand order repaired and checked by reading, not compiled, not run on hardware
//
#include <hip/hip_runtime.h>
#include <hip/hip_bf16.h>
#include <math.h>

typedef __attribute__((ext_vector_type(2)))  float  v2f;
typedef __attribute__((ext_vector_type(8)))  float  v8f;
typedef __attribute__((ext_vector_type(16))) __bf16 v16bf;

union BF16Frag { unsigned int u[8]; v16bf v; };

#define GF_ACC  1
#define GF_RELU 2
#define GF_BF16 4

// ---------------------------------------------------------------------------
// Tiled WMMA GEMM (f32 path): C = act( A[MxK] @ B[KxN] (+C) (+bias) )
// Block: 256 threads (8 waves), tile 64x64, K-step 16, WMMA f32 16x16x4.
// ---------------------------------------------------------------------------
__global__ __launch_bounds__(256)
void gemm_wmma_f32(const float* __restrict__ A, int lda,
                   const float* __restrict__ B, int ldb,
                   void* __restrict__ Cout, int ldc,
                   const float* __restrict__ bias,
                   int M, int N, int K, int flags)
{
    __shared__ float As[16][64];
    __shared__ float Bs[16][65];

    const int tid  = threadIdx.x;
    const int wave = tid >> 5;
    const int lane = tid & 31;
    const int m0 = blockIdx.y * 64;
    const int n0 = blockIdx.x * 64;
    const int sr  = wave & 3;
    const int scb = wave >> 2;
    const int kp  = (lane >> 4) << 1;
    const int l15 = lane & 15;

    v8f acc0 = {0.f, 0.f, 0.f, 0.f, 0.f, 0.f, 0.f, 0.f};
    v8f acc1 = {0.f, 0.f, 0.f, 0.f, 0.f, 0.f, 0.f, 0.f};

    for (int k0 = 0; k0 < K; k0 += 16) {
        #pragma unroll
        for (int t = 0; t < 4; ++t) {
            int idx = tid + (t << 8);
            int kk = idx >> 6;
            int p  = idx & 63;
            int gk = k0 + kk;
            int gm = m0 + p;
            As[kk][p] = (gm < M && gk < K) ? A[(size_t)gm * lda + gk] : 0.f;
            int gn = n0 + p;
            Bs[kk][p] = (gk < K && gn < N) ? B[(size_t)gk * ldb + gn] : 0.f;
        }
        __syncthreads();

        const int mrow = (sr << 4) + l15;
        const int c0   = (scb << 4) + l15;
        const int c1   = c0 + 32;
        #pragma unroll
        for (int ks = 0; ks < 16; ks += 4) {
            v2f a, b0, b1;
            a.x  = As[ks + kp][mrow];  a.y  = As[ks + kp + 1][mrow];
            b0.x = Bs[ks + kp][c0];    b0.y = Bs[ks + kp + 1][c0];
            b1.x = Bs[ks + kp][c1];    b1.y = Bs[ks + kp + 1][c1];
            acc0 = __builtin_amdgcn_wmma_f32_16x16x4_f32(false, a, false, b0,
                                                         (short)0, acc0, false, false);
            acc1 = __builtin_amdgcn_wmma_f32_16x16x4_f32(false, a, false, b1,
                                                         (short)0, acc1, false, false);
        }
        __syncthreads();
    }

    float*          Cf = (float*)Cout;
    __hip_bfloat16* Cb = (__hip_bfloat16*)Cout;
    const int gm_base = m0 + (sr << 4) + ((lane >> 4) << 3);
    const int gc_base = n0 + (scb << 4) + l15;
    #pragma unroll
    for (int v = 0; v < 8; ++v) {
        int gm = gm_base + v;
        float vals[2] = {acc0[v], acc1[v]};
        #pragma unroll
        for (int j = 0; j < 2; ++j) {
            int gc = gc_base + j * 32;
            if (gm < M && gc < N) {
                float val = vals[j];
                size_t ci = (size_t)gm * ldc + gc;
                if (flags & GF_ACC)  val += Cf[ci];
                if (bias)            val += bias[gc];
                if (flags & GF_RELU) val = fmaxf(val, 0.f);
                if (flags & GF_BF16) Cb[ci] = __float2bfloat16(val);
                else                 Cf[ci] = val;
            }
        }
    }
}

// ---------------------------------------------------------------------------
// Tiled WMMA GEMM (bf16 path): A,B bf16 -> f32 accum, WMMA bf16 16x16x32.
// Used for the dominant per-edge-weight GEMMs (K in {64,128,256}, N mult of 64).
// LDS tiles are dword-packed; B is stored K-transposed so each lane builds its
// v16bf fragment with 32-bit LDS loads matching the ISA register layout.
// ---------------------------------------------------------------------------
__global__ __launch_bounds__(256)
void gemm_wmma_bf16(const __hip_bfloat16* __restrict__ A, int lda,
                    const __hip_bfloat16* __restrict__ B, int ldb,
                    void* __restrict__ Cout, int ldc,
                    const float* __restrict__ bias,
                    int M, int N, int K, int flags)
{
    __shared__ unsigned int AsU[64][17];   // AsU[row][kd]: K pair (2kd,2kd+1)
    __shared__ unsigned int BsU[64][17];   // BsU[col][kd]: K pair at column col

    const int tid  = threadIdx.x;
    const int wave = tid >> 5;
    const int lane = tid & 31;
    const int m0 = blockIdx.y * 64;
    const int n0 = blockIdx.x * 64;
    const int sr  = wave & 3;
    const int scb = wave >> 2;
    const int h   = lane >> 4;
    const int l15 = lane & 15;

    v8f acc0 = {0.f, 0.f, 0.f, 0.f, 0.f, 0.f, 0.f, 0.f};
    v8f acc1 = {0.f, 0.f, 0.f, 0.f, 0.f, 0.f, 0.f, 0.f};

    for (int k0 = 0; k0 < K; k0 += 32) {
        // A tile: 64 rows x 16 dwords (dword = 2 consecutive K, K even).
        #pragma unroll
        for (int t = 0; t < 4; ++t) {
            int idx = tid + (t << 8);
            int kd  = idx & 15;
            int row = idx >> 4;
            int gm = m0 + row;
            int gk = k0 + (kd << 1);
            AsU[row][kd] = (gm < M && gk < K)
                               ? *(const unsigned int*)&A[(size_t)gm * lda + gk]
                               : 0u;
        }
        // B tile (transposed into col-major): two 16-bit loads packed per dword.
        #pragma unroll
        for (int t = 0; t < 4; ++t) {
            int idx = tid + (t << 8);
            int col = idx & 63;
            int kd  = idx >> 6;
            int gn = n0 + col;
            int gk = k0 + (kd << 1);
            unsigned int lo = 0, hi = 0;
            if (gn < N && gk < K)
                lo = *(const unsigned short*)&B[(size_t)gk * ldb + gn];
            if (gn < N && gk + 1 < K)
                hi = *(const unsigned short*)&B[(size_t)(gk + 1) * ldb + gn];
            BsU[col][kd] = lo | (hi << 16);
        }
        __syncthreads();

        const int row = (sr << 4) + l15;
        const int c0  = (scb << 4) + l15;
        const int c1  = c0 + 32;
        BF16Frag a, b0, b1;
        // A v16bf: elems 0..7 -> K = 8h + i ; elems 8..15 -> K = 16 + 8h + i
        #pragma unroll
        for (int p = 0; p < 4; ++p) {
            a.u[p]     = AsU[row][(h << 2) + p];
            a.u[4 + p] = AsU[row][8 + (h << 2) + p];
        }
        // B v16bf: elems 0..7 -> K = 8h + i ; elems 8..15 -> K = 16 + 8h + i
        #pragma unroll
        for (int p = 0; p < 4; ++p) {
            b0.u[p]     = BsU[c0][(h << 2) + p];
            b0.u[4 + p] = BsU[c0][8 + (h << 2) + p];
            b1.u[p]     = BsU[c1][(h << 2) + p];
            b1.u[4 + p] = BsU[c1][8 + (h << 2) + p];
        }
        acc0 = __builtin_amdgcn_wmma_f32_16x16x32_bf16(false, a.v, false, b0.v,
                                                       (short)0, acc0, false, false);
        acc1 = __builtin_amdgcn_wmma_f32_16x16x32_bf16(false, a.v, false, b1.v,
                                                       (short)0, acc1, false, false);
        __syncthreads();
    }

    float*          Cf = (float*)Cout;
    __hip_bfloat16* Cb = (__hip_bfloat16*)Cout;
    const int gm_base = m0 + (sr << 4) + (h << 3);
    const int gc_base = n0 + (scb << 4) + l15;
    #pragma unroll
    for (int v = 0; v < 8; ++v) {
        int gm = gm_base + v;
        float vals[2] = {acc0[v], acc1[v]};
        #pragma unroll
        for (int j = 0; j < 2; ++j) {
            int gc = gc_base + j * 32;
            if (gm < M && gc < N) {
                float val = vals[j];
                size_t ci = (size_t)gm * ldc + gc;
                if (flags & GF_ACC)  val += Cf[ci];
                if (bias)            val += bias[gc];
                if (flags & GF_RELU) val = fmaxf(val, 0.f);
                if (flags & GF_BF16) Cb[ci] = __float2bfloat16(val);
                else                 Cf[ci] = val;
            }
        }
    }
}

// f32 -> bf16 conversion (for e1 weight matrices).
__global__ __launch_bounds__(256)
void cvt_bf16_k(const float* __restrict__ S, __hip_bfloat16* __restrict__ D,
                size_t n)
{
    size_t i = (size_t)blockIdx.x * 256 + threadIdx.x;
    if (i < n) D[i] = __float2bfloat16(S[i]);
}

// ---------------------------------------------------------------------------
// Deterministic neighbor-mean on per-face r x r grid (4-neighbor).
// ---------------------------------------------------------------------------
__global__ __launch_bounds__(256)
void neigh_mean_k(const float* __restrict__ X, float* __restrict__ XN,
                  int r, int c, int total)
{
    int idx = blockIdx.x * blockDim.x + threadIdx.x;
    if (idx >= total) return;
    int node = idx / c;
    int ch   = idx - node * c;
    int rr   = r * r;
    int rem  = node % rr;
    int y = rem / r;
    int x = rem - y * r;
    float s = 0.f;
    int cnt = 0;
    if (x > 0)     { s += X[(size_t)(node - 1) * c + ch]; ++cnt; }
    if (x < r - 1) { s += X[(size_t)(node + 1) * c + ch]; ++cnt; }
    if (y > 0)     { s += X[(size_t)(node - r) * c + ch]; ++cnt; }
    if (y < r - 1) { s += X[(size_t)(node + r) * c + ch]; ++cnt; }
    XN[idx] = s / (float)cnt;
}

// 2x2 average pool per face: (6*r*r, c) -> (6*(r/2)^2, c)
__global__ __launch_bounds__(256)
void pool2_k(const float* __restrict__ X, float* __restrict__ Y, int r, int c)
{
    int r2 = r >> 1;
    int total = 6 * r2 * r2 * c;
    int idx = blockIdx.x * blockDim.x + threadIdx.x;
    if (idx >= total) return;
    int node = idx / c;
    int ch   = idx - node * c;
    int rr2 = r2 * r2;
    int f   = node / rr2;
    int rem = node - f * rr2;
    int y = rem / r2;
    int x = rem - y * r2;
    size_t base = (size_t)(f * r * r + (2 * y) * r + 2 * x);
    float s = X[base * c + ch] + X[(base + 1) * c + ch] +
              X[(base + r) * c + ch] + X[(base + r + 1) * c + ch];
    Y[idx] = 0.25f * s;
}

// ConvTranspose2d(c,c,2,stride=2) per face, writes with row stride out_ld.
__global__ __launch_bounds__(256)
void upsample_k(const float* __restrict__ X, const float* __restrict__ Kk,
                const float* __restrict__ bias, float* __restrict__ Y,
                int r, int c, int out_ld)
{
    int R = 2 * r;
    int total = 6 * R * R * c;
    int idx = blockIdx.x * blockDim.x + threadIdx.x;
    if (idx >= total) return;
    int node = idx / c;
    int d    = idx - node * c;
    int RR  = R * R;
    int f   = node / RR;
    int rem = node - f * RR;
    int Yr = rem / R;
    int Xc = rem - Yr * R;
    int i = Yr >> 1, a = Yr & 1;
    int j = Xc >> 1, b = Xc & 1;
    const float* xin = X + (size_t)(f * r * r + i * r + j) * c;
    const float* kp  = Kk + (size_t)d * 4 + a * 2 + b;   // k[cin][cout][a][b]
    float s = bias[d];
    for (int cc = 0; cc < c; ++cc) s = fmaf(xin[cc], kp[(size_t)cc * c * 4], s);
    Y[(size_t)node * out_ld + d] = s;
}

// Column-slice copy (for skip-connection concat).
__global__ __launch_bounds__(256)
void copy_cols_k(const float* __restrict__ S, int lds_, float* __restrict__ D,
                 int ldd, int n, int c)
{
    int idx = blockIdx.x * blockDim.x + threadIdx.x;
    if (idx >= n * c) return;
    int node = idx / c;
    int ch   = idx - node * c;
    D[(size_t)node * ldd + ch] = S[(size_t)node * lds_ + ch];
}

// ---------------------------------------------------------------------------
// MPNN message pass: out[n,o] = relu( sum_{e->n} node[src_e] @ W_e + nnb )
// W bf16 (E x hid x hid). Edge indices reconstructed analytically from the
// reference's edge generation order. Deterministic summation order.
// ---------------------------------------------------------------------------
__global__ void mpnn_msg_k(const float* __restrict__ node,
                           const __hip_bfloat16* __restrict__ W,
                           const float* __restrict__ nnb,
                           float* __restrict__ out,
                           int r, int hid)
{
    extern __shared__ float svec[];
    int nid = blockIdx.x;
    int o   = threadIdx.x;
    int rr  = r * r;
    int f   = nid / rr;
    int rem = nid - f * rr;
    int y = rem / r;
    int x = rem - y * r;
    int G = r * (r - 1);
    int ebase = f * 4 * G;
    size_t hh = (size_t)hid * hid;

    int sidx[4], eidx[4], valid[4];
    valid[0] = (x > 0);     sidx[0] = nid - 1; eidx[0] = ebase + 0 * G + y * (r - 1) + (x - 1);
    valid[1] = (x < r - 1); sidx[1] = nid + 1; eidx[1] = ebase + 1 * G + y * (r - 1) + x;
    valid[2] = (y > 0);     sidx[2] = nid - r; eidx[2] = ebase + 2 * G + (y - 1) * r + x;
    valid[3] = (y < r - 1); sidx[3] = nid + r; eidx[3] = ebase + 3 * G + y * r + x;

    float acc = 0.f;
    for (int d = 0; d < 4; ++d) {
        if (!valid[d]) continue;                 // block-uniform branch
        __syncthreads();
        svec[o] = node[(size_t)sidx[d] * hid + o];
        __syncthreads();
        const __hip_bfloat16* w = W + (size_t)eidx[d] * hh + o;
        float a = 0.f;
        for (int i = 0; i < hid; ++i)
            a = fmaf(svec[i], __bfloat162float(w[(size_t)i * hid]), a);
        acc += a;
    }
    out[(size_t)nid * hid + o] = fmaxf(acc + nnb[o], 0.f);
}

// GRU gate update (in-place hidden update).
__global__ __launch_bounds__(256)
void gru_k(const float* __restrict__ gi, const float* __restrict__ gh,
           float* __restrict__ hidden, int n, int h)
{
    int idx = blockIdx.x * blockDim.x + threadIdx.x;
    if (idx >= n * h) return;
    int nid = idx / h;
    int o   = idx - nid * h;
    const float* a = gi + (size_t)nid * 3 * h;
    const float* b = gh + (size_t)nid * 3 * h;
    float rg = 1.f / (1.f + expf(-(a[o] + b[o])));
    float zg = 1.f / (1.f + expf(-(a[h + o] + b[h + o])));
    float cd = tanhf(a[2 * h + o] + rg * b[2 * h + o]);
    float hv = hidden[idx];
    hidden[idx] = (1.f - zg) * cd + zg * hv;
}

// ---------------------------------------------------------------------------
// Host orchestration
// ---------------------------------------------------------------------------
extern "C" void kernel_launch(void* const* d_in, const int* in_sizes, int n_in,
                              void* d_out, int out_size, void* d_ws, size_t ws_size,
                              hipStream_t stream)
{
    (void)in_sizes; (void)n_in; (void)out_size; (void)ws_size;
    auto F = [&](int i) { return (const float*)d_in[i]; };

    // --- workspace bump allocator (~252 MB total) ---
    char* wp = (char*)d_ws;
    size_t off = 0;
    auto alloc = [&](size_t elems, size_t esz) {
        void* p = wp + off;
        off += (elems * esz + 255) & ~(size_t)255;
        return p;
    };
    float* XN  = (float*)alloc((size_t)24576 * 32, 4);
    float* A1  = (float*)alloc((size_t)24576 * 32, 4);
    float* A2  = (float*)alloc((size_t)24576 * 32, 4);
    float* h22 = (float*)alloc((size_t)24576 * 16, 4);
    float* h33 = (float*)alloc((size_t)6144 * 32, 4);
    float* h44 = (float*)alloc((size_t)1536 * 64, 4);
    float* h55 = (float*)alloc((size_t)384 * 128, 4);
    float* T0  = (float*)alloc((size_t)1536 * 256, 4);
    float* HID = (float*)alloc((size_t)1536 * 256, 4);
    float* MSG = (float*)alloc((size_t)1536 * 256, 4);
    float* GI  = (float*)alloc((size_t)1536 * 768, 4);
    float* GH  = (float*)alloc((size_t)1536 * 768, 4);
    __hip_bfloat16* EHb  = (__hip_bfloat16*)alloc((size_t)5760 * 256, 2);
    __hip_bfloat16* E1WB = (__hip_bfloat16*)alloc((size_t)256 * 65536, 2);
    __hip_bfloat16* WB   = (__hip_bfloat16*)alloc((size_t)5760 * 16384, 2);

    auto gemm = [&](const float* A, int lda, const float* B, int ldb,
                    void* C, int ldc, const float* bias,
                    int M, int N, int K, int flags) {
        dim3 g((N + 63) / 64, (M + 63) / 64);
        gemm_wmma_f32<<<g, dim3(256), 0, stream>>>(A, lda, B, ldb, C, ldc,
                                                   bias, M, N, K, flags);
    };
    auto gemm_bf = [&](const __hip_bfloat16* A, int lda, const __hip_bfloat16* B,
                       int ldb, void* C, int ldc, const float* bias,
                       int M, int N, int K, int flags) {
        dim3 g((N + 63) / 64, (M + 63) / 64);
        gemm_wmma_bf16<<<g, dim3(256), 0, stream>>>(A, lda, B, ldb, C, ldc,
                                                    bias, M, N, K, flags);
    };
    auto sage = [&](const float* X, int cin, int cout, int r, int n, int wi,
                    float* Y, bool relu) {
        int nt = n * cin;
        neigh_mean_k<<<(nt + 255) / 256, 256, 0, stream>>>(X, XN, r, cin, nt);
        gemm(X,  cin, F(wi),     cout, Y, cout, nullptr,   n, cout, cin, 0);
        gemm(XN, cin, F(wi + 1), cout, Y, cout, F(wi + 2), n, cout, cin,
             GF_ACC | (relu ? GF_RELU : 0));
    };
    auto upsample = [&](const float* X, int r, int c, int ki, float* Y, int out_ld) {
        int nt = 6 * (2 * r) * (2 * r) * c;
        upsample_k<<<(nt + 255) / 256, 256, 0, stream>>>(X, F(ki), F(ki + 1),
                                                         Y, r, c, out_ld);
    };
    auto ccopy = [&](const float* S, int lds_, float* D, int ldd, int n, int c) {
        copy_cols_k<<<(n * c + 255) / 256, 256, 0, stream>>>(S, lds_, D, ldd, n, c);
    };
    auto mpnn = [&](int r, int Nn, int E, int cin, int hid, int cout,
                    const float* x, const float* ef, int pb, float* out) {
        const float *p0w = F(pb + 0),  *p0b = F(pb + 1);
        const float *p1w = F(pb + 2),  *p1b = F(pb + 3);
        const float *e0w = F(pb + 4),  *e0b = F(pb + 5);
        const float *e1w = F(pb + 6),  *e1b = F(pb + 7);
        const float *nnb = F(pb + 8);
        const float *gwih = F(pb + 9),  *gwhh = F(pb + 10);
        const float *gbih = F(pb + 11), *gbhh = F(pb + 12);
        const float *d0w = F(pb + 13), *d0b = F(pb + 14);
        const float *d1w = F(pb + 15), *d1b = F(pb + 16);
        int hh = hid * hid;
        // node projection
        gemm(x,  cin, p0w, hid, T0,  hid, p0b, Nn, hid, cin, GF_RELU);
        gemm(T0, hid, p1w, hid, HID, hid, p1b, Nn, hid, hid, 0);
        // edge network: EH (bf16) = relu(ef@e0w+e0b); W (bf16) = EH@e1w+e1b
        gemm(ef, 16, e0w, hid, EHb, hid, e0b, E, hid, 16, GF_RELU | GF_BF16);
        size_t e1n = (size_t)hid * hh;
        cvt_bf16_k<<<(unsigned)((e1n + 255) / 256), 256, 0, stream>>>(e1w, E1WB, e1n);
        gemm_bf(EHb, hid, E1WB, hh, WB, hh, e1b, E, hh, hid, GF_BF16);
        for (int t = 0; t < 3; ++t) {
            mpnn_msg_k<<<Nn, hid, hid * sizeof(float), stream>>>(HID, WB, nnb,
                                                                 MSG, r, hid);
            gemm(MSG, hid, gwih, 3 * hid, GI, 3 * hid, gbih, Nn, 3 * hid, hid, 0);
            gemm(HID, hid, gwhh, 3 * hid, GH, 3 * hid, gbhh, Nn, 3 * hid, hid, 0);
            gru_k<<<(Nn * hid + 255) / 256, 256, 0, stream>>>(GI, GH, HID, Nn, hid);
        }
        gemm(HID, hid, d0w, hid,  T0,  hid,  d0b, Nn, hid,  hid, GF_RELU);
        gemm(T0,  hid, d1w, cout, out, cout, d1b, Nn, cout, hid, 0);
    };

    const int N1 = 24576, N2 = 6144, N3 = 1536, N4 = 384, N5 = 96;
    const int E3 = 5760, E4 = 1344, E5 = 288;
    const float* in_feat = F(0);
    const float* edge3 = F(1);
    const float* edge4 = F(2);
    const float* edge5 = F(3);

    // ---- encoder ----
    sage(in_feat, 8, 16, 64, N1, 14, A1, true);                // conv1
    sage(A1, 16, 16, 64, N1, 17, h22, true);                   // conv2 (skip)
    pool2_k<<<(N2 * 16 + 255) / 256, 256, 0, stream>>>(h22, A1, 64, 16);
    sage(A1, 16, 32, 32, N2, 20, A2, true);                    // conv3
    sage(A2, 32, 32, 32, N2, 23, h33, true);                   // conv33 (skip)
    pool2_k<<<(N3 * 32 + 255) / 256, 256, 0, stream>>>(h33, A1, 32, 32);
    sage(A1, 32, 64, 16, N3, 26, A2, true);                    // conv4
    sage(A2, 64, 64, 16, N3, 29, h44, true);                   // conv44 (skip)
    pool2_k<<<(N4 * 64 + 255) / 256, 256, 0, stream>>>(h44, A1, 16, 64);
    mpnn(8, N4, E4, 64, 64, 128, A1, edge4, 32, A2);           // MP1
    sage(A2, 128, 128, 8, N4, 49, h55, true);                  // conv55 (skip)
    pool2_k<<<(N5 * 128 + 255) / 256, 256, 0, stream>>>(h55, A1, 8, 128);
    mpnn(4, N5, E5, 128, 128, 256, A1, edge5, 52, A2);         // MP2

    // ---- bottleneck ----
    sage(A2, 256, 256, 4, N5, 69, A1, true);                   // conv66
    sage(A1, 256, 128, 4, N5, 72, A2, true);                   // conv7

    // ---- decoder ----
    upsample(A2, 4, 128, 133, A1, 256);                        // up1 -> cols [0,128)
    ccopy(h55, 128, A1 + 128, 256, N4, 128);                   // concat h55
    mpnn(8, N4, E4, 256, 256, 128, A1, edge4, 75, A2);         // MP3
    sage(A2, 128, 128, 8, N4, 92, A1, true);                   // conv77
    sage(A1, 128, 64, 8, N4, 95, A2, true);                    // conv8
    upsample(A2, 8, 64, 135, A1, 128);                         // up2
    ccopy(h44, 64, A1 + 64, 128, N3, 64);                      // concat h44
    mpnn(16, N3, E3, 128, 128, 64, A1, edge3, 98, A2);         // MP4
    sage(A2, 64, 64, 16, N3, 115, A1, true);                   // conv88
    sage(A1, 64, 32, 16, N3, 118, A2, true);                   // conv9
    upsample(A2, 16, 32, 137, A1, 64);                         // up3
    ccopy(h33, 32, A1 + 32, 64, N2, 32);                       // concat h33
    sage(A1, 64, 32, 32, N2, 118, A2, true);                   // conv9 (shared)
    sage(A2, 32, 32, 32, N2, 121, A1, true);                   // conv99
    sage(A1, 32, 16, 32, N2, 124, A2, true);                   // conv10
    upsample(A2, 32, 16, 139, A1, 32);                         // up4
    ccopy(h22, 16, A1 + 16, 32, N1, 16);                       // concat h22
    sage(A1, 32, 16, 64, N1, 124, A2, true);                   // conv10 (shared)
    sage(A2, 16, 16, 64, N1, 127, A1, true);                   // conv101
    sage(A1, 16, 8, 64, N1, 130, (float*)d_out, false);        // conv11 -> out
}
